// GLmMoeDsaIndexer_58806692217353
// MI455X (gfx1250) — matrix operand order repaired and checked by reading, not compiled, not run on hardware
//
#include <hip/hip_runtime.h>
#include <hip/hip_bf16.h>

// ---------------------------------------------------------------------------
// Problem constants
// ---------------------------------------------------------------------------
#define S_LEN   2048
#define HIDDEN  2048
#define NHEADS  32
#define HDIM    128
#define ROPE    64
#define TOPK    512
#define QLORA   1536
#define QOUT    (NHEADS * HDIM)   // 4096

typedef _Float16 v16h __attribute__((ext_vector_type(16)));
typedef _Float16 v8h  __attribute__((ext_vector_type(8)));
typedef float    v8f  __attribute__((ext_vector_type(8)));

// ---------------------------------------------------------------------------
// Prep: f32 -> f16 copy (q_resid)
// ---------------------------------------------------------------------------
__global__ void cvt_f16_kernel(const float* __restrict__ in,
                               _Float16* __restrict__ out, int n) {
    int i = blockIdx.x * blockDim.x + threadIdx.x;
    if (i < n) out[i] = (_Float16)in[i];
}

// Prep: Wq_b [QLORA][QOUT] f32 -> wqbT [QOUT][QLORA] f16 (transposed for B-frag loads)
__global__ void transpose_wqb_kernel(const float* __restrict__ wqb,
                                     _Float16* __restrict__ wqbT) {
    int e = blockIdx.x * blockDim.x + threadIdx.x;
    if (e >= QOUT * QLORA) return;
    int n = e / QLORA;
    int k = e - n * QLORA;
    wqbT[e] = (_Float16)wqb[k * QOUT + n];
}

// ---------------------------------------------------------------------------
// K path: k = LN(hidden @ Wk) * gamma + beta, RoPE(interleave) on first 64 dims
// one block per token t, 128 threads (one per output dim)
// ---------------------------------------------------------------------------
__global__ void kproj_kernel(const float* __restrict__ hidden,
                             const float* __restrict__ Wk,
                             const float* __restrict__ gamma,
                             const float* __restrict__ beta,
                             const float* __restrict__ cosv,
                             const float* __restrict__ sinv,
                             _Float16* __restrict__ kf16) {
    int t = blockIdx.x;
    int d = threadIdx.x;                 // 0..127
    const float* hrow = hidden + t * HIDDEN;
    float acc = 0.f;
    for (int i = 0; i < HIDDEN; ++i)
        acc = fmaf(hrow[i], Wk[i * HDIM + d], acc);

    __shared__ float r1[HDIM];
    __shared__ float r2[HDIM];
    r1[d] = acc;
    r2[d] = acc * acc;
    __syncthreads();
    for (int off = HDIM / 2; off > 0; off >>= 1) {
        if (d < off) { r1[d] += r1[d + off]; r2[d] += r2[d + off]; }
        __syncthreads();
    }
    float mean = r1[0] * (1.0f / HDIM);
    float var  = r2[0] * (1.0f / HDIM) - mean * mean;
    float inv  = rsqrtf(var + 1e-5f);
    float kn   = (acc - mean) * inv * gamma[d] + beta[d];

    __shared__ float kln[HDIM];
    kln[d] = kn;
    __syncthreads();

    float outv = kn;
    if (d < ROPE) {
        int   j  = d & 31;
        float c  = cosv[t * ROPE + d];
        float si = sinv[t * ROPE + d];
        float x0 = kln[2 * j];
        float x1 = kln[2 * j + 1];
        outv = (d < 32) ? (x0 * c - x1 * si) : (x1 * c + x0 * si);
    }
    kf16[t * HDIM + d] = (_Float16)outv;
}

// ---------------------------------------------------------------------------
// Q GEMM: qf16[2048][4096] = q_residf16[2048][1536] @ wqbT^T, f16 WMMA
// one wave per 16x64 output strip (4 accumulators, A-frag reused 4x).
// All 4 B-frags batched before the 4 WMMAs -> one clause + one wait per K-step.
// grid = (QOUT/64, S_LEN/16)
// ---------------------------------------------------------------------------
__global__ void qgemm_kernel(const _Float16* __restrict__ A,   // [S][QLORA]
                             const _Float16* __restrict__ BT,  // [QOUT][QLORA]
                             _Float16* __restrict__ C) {       // [S][QOUT]
    int lane = threadIdx.x;
    int n0 = blockIdx.x * 64;
    int m0 = blockIdx.y * 16;
    int lo = lane & 15, hi = lane >> 4;

    v8f c[4];
    for (int tt = 0; tt < 4; ++tt)
        for (int i = 0; i < 8; ++i) c[tt][i] = 0.f;

    const _Float16* arow  = A + (m0 + lo) * QLORA;
    const _Float16* brow0 = BT + (n0 + lo) * QLORA;          // tt stride = 16*QLORA

    for (int kk = 0; kk < QLORA / 32; ++kk) {
        int abase = kk * 32 + hi * 16;
        int bbase = kk * 32 + hi * 16;
        union { v16h v; v8h h2[2]; } af;
        af.h2[0] = *(const v8h*)(arow + abase);
        af.h2[1] = *(const v8h*)(arow + abase + 8);
        v16h bf[4];
#pragma unroll
        for (int tt = 0; tt < 4; ++tt)
            bf[tt] = *(const v16h*)(brow0 + tt * 16 * QLORA + bbase);
#pragma unroll
        for (int tt = 0; tt < 4; ++tt)
            c[tt] = __builtin_amdgcn_wmma_f32_16x16x32_f16(false, af.v, false, bf[tt],
                                                           (short)0, c[tt], false, false);
    }
#pragma unroll
    for (int tt = 0; tt < 4; ++tt) {
        _Float16* crow = C + n0 + tt * 16 + lo;
        for (int r = 0; r < 8; ++r)
            crow[(m0 + r + 8 * hi) * QOUT] = (_Float16)c[tt][r];
    }
}

// ---------------------------------------------------------------------------
// In-place interleaved RoPE on q: per row s, per head, first 64 dims
// one block per s, 256 threads; stage rope region in LDS (read-before-write)
// ---------------------------------------------------------------------------
__global__ void qrope_kernel(_Float16* __restrict__ qf16,
                             const float* __restrict__ cosv,
                             const float* __restrict__ sinv) {
    int s = blockIdx.x;
    int tid = threadIdx.x;
    __shared__ float ri[NHEADS * ROPE];   // 2048 floats = 8KB
    _Float16* row = qf16 + s * QOUT;

    for (int e = tid; e < NHEADS * ROPE; e += 256) {
        int h = e >> 6, d = e & 63;
        ri[e] = (float)row[h * HDIM + d];
    }
    __syncthreads();
    for (int e = tid; e < NHEADS * ROPE; e += 256) {
        int h = e >> 6, d = e & 63;
        int j = d & 31;
        float c  = cosv[s * ROPE + d];
        float si = sinv[s * ROPE + d];
        float x0 = ri[h * ROPE + 2 * j];
        float x1 = ri[h * ROPE + 2 * j + 1];
        float o  = (d < 32) ? (x0 * c - x1 * si) : (x1 * c + x0 * si);
        row[h * HDIM + d] = (_Float16)o;
    }
}

// ---------------------------------------------------------------------------
// Scores: scores[s][t] = sum_h relu( q[s,h,:] . k[t,:] ) + mask[s][t]
// one wave per 16x64 (s,t) strip; K strip staged in LDS (16KB), B-frags via DS
// batched ahead of the 4 WMMAs; A-frag reused across 4 accumulators.
// ---------------------------------------------------------------------------
__global__ void score_kernel(const _Float16* __restrict__ qf16, // [S][NHEADS*HDIM]
                             const _Float16* __restrict__ kf16, // [S][HDIM]
                             const float* __restrict__ mask,    // [S][S]
                             float* __restrict__ scores) {      // [S][S]
    int lane = threadIdx.x;
    int t0 = blockIdx.x * 64;
    int s0 = blockIdx.y * 16;
    int lo = lane & 15, hi = lane >> 4;

    // Stage K strip [64][128] f16 -> LDS (16KB)
    __shared__ alignas(32) _Float16 kt[64 * HDIM];
    {
        const _Float16* src = kf16 + t0 * HDIM;
        for (int i = lane; i < (64 * HDIM) / 8; i += 32)
            *(v8h*)(kt + i * 8) = *(const v8h*)(src + i * 8);
    }
    __syncthreads();

    float sacc[4][8];
    for (int tt = 0; tt < 4; ++tt)
        for (int r = 0; r < 8; ++r) sacc[tt][r] = 0.f;

    const _Float16* qrow = qf16 + (s0 + lo) * QOUT;

    for (int h = 0; h < NHEADS; ++h) {
        const _Float16* qh = qrow + h * HDIM;
        __builtin_prefetch(qh + HDIM, 0, 0);   // global_prefetch_b8 for next head
        v8f c[4];
        for (int tt = 0; tt < 4; ++tt)
            for (int i = 0; i < 8; ++i) c[tt][i] = 0.f;
#pragma unroll
        for (int kk = 0; kk < HDIM / 32; ++kk) {
            int abase = kk * 32 + hi * 16;
            int bbase = kk * 32 + hi * 16;
            union { v16h v; v8h h2[2]; } af;
            af.h2[0] = *(const v8h*)(qh + abase);
            af.h2[1] = *(const v8h*)(qh + abase + 8);
            v16h bf[4];
#pragma unroll
            for (int tt = 0; tt < 4; ++tt)
                bf[tt] = *(const v16h*)(kt + (tt * 16 + lo) * HDIM + bbase);
#pragma unroll
            for (int tt = 0; tt < 4; ++tt)
                c[tt] = __builtin_amdgcn_wmma_f32_16x16x32_f16(false, af.v, false, bf[tt],
                                                               (short)0, c[tt], false, false);
        }
        for (int tt = 0; tt < 4; ++tt)
            for (int r = 0; r < 8; ++r) sacc[tt][r] += fmaxf(c[tt][r], 0.f);
    }
#pragma unroll
    for (int tt = 0; tt < 4; ++tt) {
        for (int r = 0; r < 8; ++r) {
            int srow = s0 + r + 8 * hi;
            int tcol = t0 + tt * 16 + lo;
            scores[srow * S_LEN + tcol] = sacc[tt][r] + mask[srow * S_LEN + tcol];
        }
    }
}

// ---------------------------------------------------------------------------
// Top-K: per row s, bitonic sort (value desc, index asc) in LDS, emit 512 idx
// ---------------------------------------------------------------------------
__global__ void topk_kernel(const float* __restrict__ scores,
                            int* __restrict__ out) {
    int s = blockIdx.x;
    int tid = threadIdx.x;
    __shared__ float v[S_LEN];
    __shared__ int   id[S_LEN];
    for (int i = tid; i < S_LEN; i += 256) {
        v[i] = scores[s * S_LEN + i];
        id[i] = i;
    }
    __syncthreads();
    for (int k = 2; k <= S_LEN; k <<= 1) {
        for (int j = k >> 1; j > 0; j >>= 1) {
            for (int i = tid; i < S_LEN; i += 256) {
                int ixj = i ^ j;
                if (ixj > i) {
                    bool desc = ((i & k) == 0);
                    float va = v[i], vb = v[ixj];
                    int   ia = id[i], ib = id[ixj];
                    bool aBefore = (va > vb) || (va == vb && ia < ib);
                    if (desc ? !aBefore : aBefore) {
                        v[i] = vb; v[ixj] = va;
                        id[i] = ib; id[ixj] = ia;
                    }
                }
            }
            __syncthreads();
        }
    }
    for (int i = tid; i < TOPK; i += 256) out[s * TOPK + i] = id[i];
}

// ---------------------------------------------------------------------------
// Launcher
// ---------------------------------------------------------------------------
extern "C" void kernel_launch(void* const* d_in, const int* in_sizes, int n_in,
                              void* d_out, int out_size, void* d_ws, size_t ws_size,
                              hipStream_t stream) {
    const float* hidden  = (const float*)d_in[0];
    const float* q_resid = (const float*)d_in[1];
    const float* cosv    = (const float*)d_in[2];
    const float* sinv    = (const float*)d_in[3];
    const float* mask    = (const float*)d_in[4];
    const float* Wq_b    = (const float*)d_in[5];
    const float* Wk      = (const float*)d_in[6];
    const float* gamma   = (const float*)d_in[7];
    const float* beta    = (const float*)d_in[8];
    // d_in[9] = Ww, computed-then-discarded in the reference; unused.
    int* out = (int*)d_out;

    // Workspace layout (all offsets 256B-aligned)
    char* ws = (char*)d_ws;
    size_t off = 0;
    _Float16* q_residf16 = (_Float16*)(ws + off); off += (size_t)S_LEN * QLORA * 2;  // 6 MB
    _Float16* wqbT       = (_Float16*)(ws + off); off += (size_t)QOUT * QLORA * 2;   // 12 MB
    _Float16* qf16       = (_Float16*)(ws + off); off += (size_t)S_LEN * QOUT * 2;   // 16 MB
    _Float16* kf16       = (_Float16*)(ws + off); off += (size_t)S_LEN * HDIM * 2;   // 0.5 MB
    float*    scores     = (float*)   (ws + off);                                    // 16 MB

    // 1. precision conversion + weight transpose
    {
        int n = S_LEN * QLORA;
        cvt_f16_kernel<<<(n + 255) / 256, 256, 0, stream>>>(q_resid, q_residf16, n);
        int m = QOUT * QLORA;
        transpose_wqb_kernel<<<(m + 255) / 256, 256, 0, stream>>>(Wq_b, wqbT);
    }
    // 2. K path: proj + LN + RoPE
    kproj_kernel<<<S_LEN, HDIM, 0, stream>>>(hidden, Wk, gamma, beta, cosv, sinv, kf16);
    // 3. Q path: WMMA GEMM (16x64 strips), then in-place RoPE
    qgemm_kernel<<<dim3(QOUT / 64, S_LEN / 16), 32, 0, stream>>>(q_residf16, wqbT, qf16);
    qrope_kernel<<<S_LEN, 256, 0, stream>>>(qf16, cosv, sinv);
    // 4. fused score GEMM (relu + head-sum + mask), K strip in LDS
    score_kernel<<<dim3(S_LEN / 64, S_LEN / 16), 32, 0, stream>>>(qf16, kf16, mask, scores);
    // 5. per-row top-512
    topk_kernel<<<S_LEN, 256, 0, stream>>>(scores, out);
}
